// TransformerDecoder_20916490732189
// MI455X (gfx1250) — matrix operand order repaired and checked by reading, not compiled, not run on hardware
//
#include <hip/hip_runtime.h>
#include <hip/hip_bf16.h>
#include <math.h>

typedef unsigned short u16;
typedef __attribute__((ext_vector_type(4)))  unsigned int u32x4;
typedef __attribute__((ext_vector_type(16))) __bf16       bf16x16;
typedef __attribute__((ext_vector_type(8)))  float        f32x8;
typedef __attribute__((__vector_size__(16))) int          v4i;   // matches async-LDS builtin param

#define DIM   1024
#define MLP_D 4096
#define BN    4096          // B*N rows
#define HEADS 16
#define HD    64
#define ATT_SCALE 0.125f    // 1/sqrt(64)

#if __has_builtin(__builtin_amdgcn_global_load_async_to_lds_b128) && \
    __has_builtin(__builtin_amdgcn_s_wait_asynccnt)
#define HAVE_ASYNC_LDS 1
#else
#define HAVE_ASYNC_LDS 0
#endif

__device__ __forceinline__ u16 f2bf(float f) {
  unsigned int u = __float_as_uint(f);
  u += 0x7FFFu + ((u >> 16) & 1u);   // round-to-nearest-even
  return (u16)(u >> 16);
}

__device__ __forceinline__ f32x8 zero8() {
  f32x8 z;
#pragma unroll
  for (int i = 0; i < 8; ++i) z[i] = 0.0f;
  return z;
}

// 16-byte global -> LDS copy: async DMA path on CDNA5, VGPR round-trip fallback.
__device__ __forceinline__ void g2l16(const u16* g, u16* l) {
#if HAVE_ASYNC_LDS
  __builtin_amdgcn_global_load_async_to_lds_b128((v4i*)g, (v4i*)l, 0, 0);
#else
  *(u32x4*)l = *(const u32x4*)g;
#endif
}

__device__ __forceinline__ void wait_async() {
#if HAVE_ASYNC_LDS
  __builtin_amdgcn_s_wait_asynccnt(0);
#endif
}

// A-matrix fragment (16x32 bf16): lane%16 = M row; lanes 0-15 hold K [k..k+8)&[k+16..k+24),
// lanes 16-31 hold K [k+8..k+16)&[k+24..k+32). Two ds_load_b128 per lane.
__device__ __forceinline__ bf16x16 ld_fragA(const u16* p, int rowBase, int stride, int kOff) {
  int lane = threadIdx.x & 31;
  const u16* base = p + (rowBase + (lane & 15)) * stride + kOff + ((lane >> 4) << 3);
  union { bf16x16 v; u32x4 u[2]; } f;
  f.u[0] = *(const u32x4*)(base);
  f.u[1] = *(const u32x4*)(base + 16);
  return f.v;
}

// B-matrix fragment (32x16 bf16): lane%16 = N col; lanes 0-15 hold K [k..k+8)&[k+16..k+24),
// lanes 16-31 hold K [k+8..k+16)&[k+24..k+32). Source tile stored [N][K] in LDS (contiguous K).
__device__ __forceinline__ bf16x16 ld_fragB(const u16* p, int colBase, int stride, int kOff) {
  int lane = threadIdx.x & 31;
  const u16* base = p + (colBase + (lane & 15)) * stride + kOff + ((lane >> 4) << 3);
  union { bf16x16 v; u32x4 u[2]; } f;
  f.u[0] = *(const u32x4*)(base);
  f.u[1] = *(const u32x4*)(base + 16);
  return f.v;
}

__device__ __forceinline__ f32x8 wmma_bf16(bf16x16 a, bf16x16 b, f32x8 c) {
  return __builtin_amdgcn_wmma_f32_16x16x32_bf16(false, a, false, b, (short)0, c, false, false);
}

// ---------------------------------------------------------------- elementwise
__global__ __launch_bounds__(256) void copy_f32_kernel(const float* __restrict__ s,
                                                       float* __restrict__ d, int n) {
  int i = blockIdx.x * 256 + threadIdx.x;
  if (i < n) d[i] = s[i];
}

// fp32 [K][N] -> bf16 transposed [N][K]  (weights converted once per layer)
__global__ __launch_bounds__(256) void cvt_t_kernel(const float* __restrict__ src,
                                                    u16* __restrict__ dst, int K, int N) {
  __shared__ u16 t[32][33];
  const int n0 = blockIdx.x * 32, k0 = blockIdx.y * 32;
  const int c = threadIdx.x & 31, r0 = threadIdx.x >> 5;   // 8 row-groups
#pragma unroll
  for (int rr = 0; rr < 4; ++rr) {
    int r = r0 + rr * 8;
    t[c][r] = f2bf(src[(size_t)(k0 + r) * N + n0 + c]);    // coalesced along n
  }
  __syncthreads();
#pragma unroll
  for (int rr = 0; rr < 4; ++rr) {
    int r = r0 + rr * 8;
    dst[(size_t)(n0 + r) * K + k0 + c] = t[r][c];          // coalesced along k
  }
}

// ---------------------------------------------------------------- layernorm
__global__ __launch_bounds__(256) void ln_kernel(const float* __restrict__ x,
                                                 const float* __restrict__ g,
                                                 const float* __restrict__ b,
                                                 u16* __restrict__ out) {
  __shared__ float red[256];
  const int row = blockIdx.x, tid = threadIdx.x;
  const float* xr = x + (size_t)row * DIM;

  float s = 0.f;
#pragma unroll
  for (int j = 0; j < 4; ++j) s += xr[tid + 256 * j];
  red[tid] = s; __syncthreads();
  for (int st = 128; st > 0; st >>= 1) { if (tid < st) red[tid] += red[tid + st]; __syncthreads(); }
  const float mean = red[0] * (1.0f / DIM);
  __syncthreads();

  float v = 0.f;
#pragma unroll
  for (int j = 0; j < 4; ++j) { float d = xr[tid + 256 * j] - mean; v += d * d; }
  red[tid] = v; __syncthreads();
  for (int st = 128; st > 0; st >>= 1) { if (tid < st) red[tid] += red[tid + st]; __syncthreads(); }
  const float rstd = rsqrtf(red[0] * (1.0f / DIM) + 1e-5f);

#pragma unroll
  for (int j = 0; j < 4; ++j) {
    int c = tid + 256 * j;
    float o = (xr[c] - mean) * rstd * g[c] + b[c];
    out[(size_t)row * DIM + c] = f2bf(o);
  }
}

// ---------------------------------------------------------------- GEMM
// C[M,N] = A[M,K](bf16) * Wt[N,K](bf16, pre-transposed); 128x128 block tile,
// 8 waves of 32x64, double-buffered LDS fed by async global->LDS copies.
// mode 0: store bf16                      (q/k projections)
// mode 1: resid = acc+bias+resid (fp32)   (wo, w2)
// mode 2: store bf16 gelu(acc+bias)       (w1)
// mode 3: store bf16 transposed [b,chan,n] (v projection; feeds attention V tile)
__global__ __launch_bounds__(256) void gemm_bf16_kernel(
    const u16* __restrict__ A, const u16* __restrict__ Wt,
    const float* __restrict__ bias, float* __restrict__ resid,
    u16* __restrict__ outB, int M, int N, int K, int mode) {
  __shared__ u16 sA[2][128 * 72];   // [m][k], k-depth 64, padded
  __shared__ u16 sB[2][128 * 72];   // [n][k]

  const int tid = threadIdx.x;
  const int wave = tid >> 5, lane = tid & 31;
  const int wm = wave >> 1, wn = wave & 1;
  const int mBase = blockIdx.y * 128, nBase = blockIdx.x * 128;

  f32x8 acc[2][4];
#pragma unroll
  for (int i = 0; i < 2; ++i)
#pragma unroll
    for (int j = 0; j < 4; ++j) acc[i][j] = zero8();

  const int row = tid >> 1, c0 = (tid & 1) << 5;   // 2 threads/row, 32 bf16 each
  const u16* gA = A  + (size_t)(mBase + row) * K + c0;
  const u16* gB = Wt + (size_t)(nBase + row) * K + c0;

  auto issue_tile = [&](int buf, int k0) {
    u16* la = &sA[buf][row * 72 + c0];
    u16* lb = &sB[buf][row * 72 + c0];
#pragma unroll
    for (int j = 0; j < 4; ++j) {
      g2l16(gA + k0 + j * 8, la + j * 8);
      g2l16(gB + k0 + j * 8, lb + j * 8);
    }
  };

  const int nIter = K >> 6;
  issue_tile(0, 0);
  for (int it = 0; it < nIter; ++it) {
    const int cur = it & 1;
    wait_async();
    __syncthreads();                       // tile `cur` visible; prev compute done
    if (it + 1 < nIter) issue_tile(cur ^ 1, (it + 1) << 6);
#pragma unroll
    for (int kk = 0; kk < 64; kk += 32) {
      bf16x16 af[2], bf[4];
#pragma unroll
      for (int fm = 0; fm < 2; ++fm) af[fm] = ld_fragA(sA[cur], wm * 32 + fm * 16, 72, kk);
#pragma unroll
      for (int fn = 0; fn < 4; ++fn) bf[fn] = ld_fragB(sB[cur], wn * 64 + fn * 16, 72, kk);
#pragma unroll
      for (int fm = 0; fm < 2; ++fm)
#pragma unroll
        for (int fn = 0; fn < 4; ++fn) acc[fm][fn] = wmma_bf16(af[fm], bf[fn], acc[fm][fn]);
    }
  }

  const int half = lane >> 4, col = lane & 15;
#pragma unroll
  for (int fm = 0; fm < 2; ++fm)
#pragma unroll
    for (int fn = 0; fn < 4; ++fn)
#pragma unroll
      for (int r = 0; r < 8; ++r) {
        int gm = mBase + wm * 32 + fm * 16 + half * 8 + r;
        int gn = nBase + wn * 64 + fn * 16 + col;
        if (gm >= M || gn >= N) continue;
        float v = acc[fm][fn][r];
        if (mode == 0) {
          outB[(size_t)gm * N + gn] = f2bf(v);
        } else if (mode == 1) {
          size_t idx = (size_t)gm * N + gn;
          resid[idx] = v + bias[gn] + resid[idx];
        } else if (mode == 2) {
          float o = v + bias[gn];
          o = 0.5f * o * (1.0f + erff(o * 0.70710678118654752f));
          outB[(size_t)gm * N + gn] = f2bf(o);
        } else { // mode 3: [b, chan, token]
          outB[((size_t)(gm >> 10) * 1024 + gn) * 1024 + (gm & 1023)] = f2bf(v);
        }
      }
}

// ---------------------------------------------------------------- flash attention
// block = 128 threads (4 waves), 64 query rows per block, 128-key tiles.
// Q/K bf16 [b, n, h*64+d] (stride DIM); V pre-transposed bf16 [b, h*64+d, n].
__global__ __launch_bounds__(128) void attn_kernel(const u16* __restrict__ Q,
                                                   const u16* __restrict__ Kb,
                                                   const u16* __restrict__ Vt,
                                                   u16* __restrict__ O) {
  __shared__ u16 sQ[64 * 72];
  __shared__ u16 sK[128 * 72];
  __shared__ u16 sVt[64 * 136];   // [d][key]
  __shared__ u16 sP[64 * 136];    // [qrow][key]

  const int tid = threadIdx.x, wave = tid >> 5, lane = tid & 31;
  const int half = lane >> 4, col = lane & 15;
  const int qt = blockIdx.x, h = blockIdx.y, b = blockIdx.z;
  const size_t base  = ((size_t)b * 1024) * DIM + h * HD;           // Q/K row-major base
  const size_t baseV = ((size_t)b * 1024 + h * HD) * 1024;          // Vt [chan][token] base

  { // Q tile 64x64 (async)
    int row = tid >> 1, d0 = (tid & 1) << 5;
    const u16* g = Q + base + (size_t)(qt * 64 + row) * DIM + d0;
    u16* l = sQ + row * 72 + d0;
#pragma unroll
    for (int j = 0; j < 4; ++j) g2l16(g + j * 8, l + j * 8);
  }

  float m_i[8], l_i[8];
  f32x8 o_acc[4];
#pragma unroll
  for (int r = 0; r < 8; ++r) { m_i[r] = -1e30f; l_i[r] = 0.0f; }
#pragma unroll
  for (int f = 0; f < 4; ++f) o_acc[f] = zero8();

  for (int kt = 0; kt < 8; ++kt) {
    { // K tile 128x64: one row per thread
      const u16* g = Kb + base + (size_t)(kt * 128 + tid) * DIM;
      u16* l = sK + tid * 72;
#pragma unroll
      for (int j = 0; j < 8; ++j) g2l16(g + j * 8, l + j * 8);
    }
    { // V tile (already [d][key]): 64 rows x 128 keys, 2 threads/row
      int row = tid >> 1, c0 = (tid & 1) << 6;
      const u16* g = Vt + baseV + (size_t)row * 1024 + kt * 128 + c0;
      u16* l = sVt + row * 136 + c0;
#pragma unroll
      for (int j = 0; j < 8; ++j) g2l16(g + j * 8, l + j * 8);
    }
    wait_async();
    __syncthreads();

    // S = Q * K^T   (wave handles 16 q rows x 128 keys)
    f32x8 s[8];
#pragma unroll
    for (int f = 0; f < 8; ++f) s[f] = zero8();
#pragma unroll
    for (int kk = 0; kk < 64; kk += 32) {
      bf16x16 aq = ld_fragA(sQ, wave * 16, 72, kk);
#pragma unroll
      for (int f = 0; f < 8; ++f)
        s[f] = wmma_bf16(aq, ld_fragB(sK, f * 16, 72, kk), s[f]);
    }

#pragma unroll
    for (int f = 0; f < 8; ++f)
#pragma unroll
      for (int r = 0; r < 8; ++r) s[f][r] *= ATT_SCALE;

    // online softmax; row = wave*16 + half*8 + r
#pragma unroll
    for (int r = 0; r < 8; ++r) {
      float mx = -1e30f;
#pragma unroll
      for (int f = 0; f < 8; ++f) mx = fmaxf(mx, s[f][r]);
#pragma unroll
      for (int d = 1; d < 16; d <<= 1) mx = fmaxf(mx, __shfl_xor(mx, d));
      float mn = fmaxf(m_i[r], mx);
      float al = expf(m_i[r] - mn);
      float rs = 0.f;
#pragma unroll
      for (int f = 0; f < 8; ++f) { float p = expf(s[f][r] - mn); s[f][r] = p; rs += p; }
#pragma unroll
      for (int d = 1; d < 16; d <<= 1) rs += __shfl_xor(rs, d);
      l_i[r] = l_i[r] * al + rs;
      m_i[r] = mn;
#pragma unroll
      for (int f = 0; f < 4; ++f) o_acc[f][r] *= al;
    }

    // P -> LDS (bf16, A-operand layout source)
#pragma unroll
    for (int f = 0; f < 8; ++f)
#pragma unroll
      for (int r = 0; r < 8; ++r)
        sP[(wave * 16 + half * 8 + r) * 136 + f * 16 + col] = f2bf(s[f][r]);
    __syncthreads();

    // O += P * V
#pragma unroll
    for (int kk = 0; kk < 128; kk += 32) {
      bf16x16 ap = ld_fragA(sP, wave * 16, 136, kk);
#pragma unroll
      for (int f = 0; f < 4; ++f)
        o_acc[f] = wmma_bf16(ap, ld_fragB(sVt, f * 16, 136, kk), o_acc[f]);
    }
    __syncthreads();
  }

#pragma unroll
  for (int f = 0; f < 4; ++f)
#pragma unroll
    for (int r = 0; r < 8; ++r) {
      int gq = qt * 64 + wave * 16 + half * 8 + r;
      float v = o_acc[f][r] / l_i[r];
      O[((size_t)b * 1024 + gq) * DIM + h * HD + f * 16 + col] = f2bf(v);
    }
}

// ---------------------------------------------------------------- orchestration
extern "C" void kernel_launch(void* const* d_in, const int* in_sizes, int n_in,
                              void* d_out, int out_size, void* d_ws, size_t ws_size,
                              hipStream_t stream) {
  const float* x1    = (const float*)d_in[0];
  const float* x2    = (const float*)d_in[1];
  const float* ln1_g = (const float*)d_in[2];
  const float* ln1_b = (const float*)d_in[3];
  const float* wq    = (const float*)d_in[4];
  const float* wk    = (const float*)d_in[5];
  const float* wv    = (const float*)d_in[6];
  const float* wo    = (const float*)d_in[7];
  const float* bo    = (const float*)d_in[8];
  const float* ln2_g = (const float*)d_in[9];
  const float* ln2_b = (const float*)d_in[10];
  const float* w1    = (const float*)d_in[11];
  const float* b1    = (const float*)d_in[12];
  const float* w2    = (const float*)d_in[13];
  const float* b2    = (const float*)d_in[14];

  float* x1buf = (float*)d_out;           // residual stream lives in d_out (fp32)

  const size_t DD = (size_t)DIM * DIM;    // 1M
  const size_t DM = (size_t)DIM * MLP_D;  // 4M
  const size_t ACT = (size_t)BN * DIM;    // 4M

  size_t ofs = 0;
  auto carve = [&](size_t elems) { u16* p = (u16*)((char*)d_ws + ofs); ofs += elems * sizeof(u16); return p; };
  u16* wqb = carve(DD); u16* wkb = carve(DD); u16* wvb = carve(DD); u16* wob = carve(DD);
  u16* w1b = carve(DM); u16* w2b = carve(DM);
  u16* n1  = carve(ACT); u16* n2  = carve(ACT);
  u16* qb  = carve(ACT); u16* kb  = carve(ACT); u16* vbT = carve(ACT);
  u16* ao  = carve(ACT);
  u16* hid = carve((size_t)BN * MLP_D);
  (void)ws_size; (void)n_in; (void)in_sizes; (void)out_size;

  copy_f32_kernel<<<(int)(ACT / 256), 256, 0, stream>>>(x1, x1buf, (int)ACT);

  for (int i = 0; i < 6; ++i) {
    // convert + transpose weights: fp32 [K][N] -> bf16 [N][K]
    cvt_t_kernel<<<dim3(32, 32),  256, 0, stream>>>(wq + i * DD, wqb, DIM, DIM);
    cvt_t_kernel<<<dim3(32, 32),  256, 0, stream>>>(wk + i * DD, wkb, DIM, DIM);
    cvt_t_kernel<<<dim3(32, 32),  256, 0, stream>>>(wv + i * DD, wvb, DIM, DIM);
    cvt_t_kernel<<<dim3(32, 32),  256, 0, stream>>>(wo + i * DD, wob, DIM, DIM);
    cvt_t_kernel<<<dim3(128, 32), 256, 0, stream>>>(w1 + i * DM, w1b, DIM, MLP_D);
    cvt_t_kernel<<<dim3(32, 128), 256, 0, stream>>>(w2 + i * DM, w2b, MLP_D, DIM);

    // PreNorm(MCA): same ln1 params on x1 and x2
    ln_kernel<<<BN, 256, 0, stream>>>(x1buf, ln1_g + i * DIM, ln1_b + i * DIM, n1);
    ln_kernel<<<BN, 256, 0, stream>>>(x2,    ln1_g + i * DIM, ln1_b + i * DIM, n2);

    gemm_bf16_kernel<<<dim3(8, 32), 256, 0, stream>>>(n1, wqb, nullptr, nullptr, qb,  BN, DIM, DIM, 0);
    gemm_bf16_kernel<<<dim3(8, 32), 256, 0, stream>>>(n2, wkb, nullptr, nullptr, kb,  BN, DIM, DIM, 0);
    gemm_bf16_kernel<<<dim3(8, 32), 256, 0, stream>>>(n2, wvb, nullptr, nullptr, vbT, BN, DIM, DIM, 3);

    attn_kernel<<<dim3(16, HEADS, 4), 128, 0, stream>>>(qb, kb, vbT, ao);

    // x1 += attn_out @ wo + bo
    gemm_bf16_kernel<<<dim3(8, 32), 256, 0, stream>>>(ao, wob, bo + i * DIM, x1buf, nullptr, BN, DIM, DIM, 1);

    // PreNorm(FFN)
    ln_kernel<<<BN, 256, 0, stream>>>(x1buf, ln2_g + i * DIM, ln2_b + i * DIM, n1);
    gemm_bf16_kernel<<<dim3(32, 32), 256, 0, stream>>>(n1, w1b, b1 + i * MLP_D, nullptr, hid, BN, MLP_D, DIM, 2);
    gemm_bf16_kernel<<<dim3(8, 32), 256, 0, stream>>>(hid, w2b, b2 + i * DIM, x1buf, nullptr, BN, DIM, MLP_D, 1);
  }
}
